// WordReadout_10428180595136
// MI455X (gfx1250) — matrix operand order repaired and checked by reading, not compiled, not run on hardware
//
#include <hip/hip_runtime.h>
#include <hip/hip_bf16.h>
#include <math.h>

typedef __bf16 bf16_t;
typedef __attribute__((ext_vector_type(16))) __bf16 v16bf;
typedef __attribute__((ext_vector_type(8)))  __bf16 v8bf;
typedef __attribute__((ext_vector_type(8)))  float  v8f;
typedef __attribute__((ext_vector_type(4)))  float  v4f;

#define H     256
#define TM    64          // rows per workgroup
#define HSTR  264         // padded LDS row stride (bf16 elems), 16B-aligned rows
#define NSEG  512

static_assert(sizeof(bf16_t) == 2, "bf16 must be 2 bytes");

#if __has_builtin(__builtin_amdgcn_global_load_async_to_lds_b32) && \
    __has_builtin(__builtin_amdgcn_s_wait_asynccnt)
#define HAVE_ASYNC_LDS 1
#else
#define HAVE_ASYNC_LDS 0
#endif

// ---------------------------------------------------------------------------
// init: zero output + counts, convert W1/W2 to bf16 in workspace
// ---------------------------------------------------------------------------
__global__ __launch_bounds__(256) void wr_init_kernel(
    const float* __restrict__ W1, const float* __restrict__ W2,
    bf16_t* __restrict__ w1b, bf16_t* __restrict__ w2b,
    float* __restrict__ cnt, float* __restrict__ out)
{
    int i = blockIdx.x * 256 + threadIdx.x;
    if (i < NSEG * 2 * H) out[i] = 0.0f;
    if (i < H * H) {
        w1b[i] = (bf16_t)W1[i];
        w2b[i] = (bf16_t)W2[i];
    }
    if (i < NSEG) cnt[i] = 0.0f;
}

// ---------------------------------------------------------------------------
__device__ __forceinline__ v16bf frag16(v8bf lo, v8bf hi) {
    return __builtin_shufflevector(lo, hi, 0,1,2,3,4,5,6,7,8,9,10,11,12,13,14,15);
}

// One 256-K GEMM pass: A (bf16) from LDS tile `hl`, B from global weights `W`
// (row-major W^T access: lane = output column, contiguous K run per lane).
__device__ __forceinline__ void gemm_pass(const bf16_t* hl, const bf16_t* __restrict__ W,
                                          int wave, int l15, int lhi, v8f acc[4][2])
{
    for (int k0 = 0; k0 < H; k0 += 32) {
        v16bf bfr[2];
        #pragma unroll
        for (int t = 0; t < 2; ++t) {
            int n = wave * 32 + t * 16 + l15;
            const v8bf* wp = (const v8bf*)(W + (size_t)n * H + k0 + lhi * 16);
            bfr[t] = frag16(wp[0], wp[1]);
            __builtin_prefetch(W + (size_t)n * H + k0 + 32, 0, 3);
        }
        #pragma unroll
        for (int m = 0; m < 4; ++m) {
            const v8bf* ap = (const v8bf*)(hl + (m * 16 + l15) * HSTR + k0 + lhi * 16);
            v16bf af = frag16(ap[0], ap[1]);   // K: kw..kw+15 contiguous
            acc[m][0] = __builtin_amdgcn_wmma_f32_16x16x32_bf16(
                false, af, false, bfr[0], (short)0, acc[m][0], false, false);
            acc[m][1] = __builtin_amdgcn_wmma_f32_16x16x32_bf16(
                false, af, false, bfr[1], (short)0, acc[m][1], false, false);
        }
    }
}

// ---------------------------------------------------------------------------
// fused: GEMM1(relu) -> GEMM2(sigmoid gate) -> segmented sum/max/count
// ---------------------------------------------------------------------------
__global__ __launch_bounds__(256, 2) void wr_fused_kernel(
    const float* __restrict__ x, const int* __restrict__ batch,
    const bf16_t* __restrict__ w1b, const float* __restrict__ b1,
    const bf16_t* __restrict__ w2b, const float* __restrict__ b2,
    float* __restrict__ out, float* __restrict__ cnt)
{
    // Aliased tile buffer: x (bf16) during GEMM1, h (bf16) during GEMM2. ~33KB.
    __shared__ __align__(16) bf16_t hl[TM * HSTR];
    __shared__ int sids[TM];

    const int tid  = threadIdx.x;
    const int wave = tid >> 5;
    const int lane = tid & 31;
    const int l15  = lane & 15;
    const int lhi  = lane >> 4;            // 0 or 1
    const int row0 = blockIdx.x * TM;

    // segment ids: async copy global->LDS, overlapped with GEMM1
#if HAVE_ASYNC_LDS
    if (tid < TM) {
        int* bsrc = const_cast<int*>(batch) + row0 + tid;
        __builtin_amdgcn_global_load_async_to_lds_b32(
            (__attribute__((address_space(1))) int*)bsrc,
            (__attribute__((address_space(3))) int*)&sids[tid], 0, 0);
    }
#else
    if (tid < TM) sids[tid] = batch[row0 + tid];
#endif

    // ---- cooperative staging: x tile fp32 -> bf16 in LDS (once per WG) ----
    #pragma unroll
    for (int i = 0; i < 8; ++i) {
        int e   = i * 2048 + tid * 8;      // 64*256 elements total
        int row = e >> 8;
        int k   = e & 255;
        const float* xp = x + (size_t)(row0 + row) * H + k;
        v4f a = *(const v4f*)xp;
        v4f b = *(const v4f*)(xp + 4);
        v8bf p;
        #pragma unroll
        for (int j = 0; j < 4; ++j) {
            p[j]     = (bf16_t)a[j];
            p[4 + j] = (bf16_t)b[j];
        }
        *(v8bf*)(hl + row * HSTR + k) = p;
    }
    __syncthreads();

    // ---------------- GEMM1: h = relu(x @ W1^T + b1) ----------------
    v8f acc[4][2];
    #pragma unroll
    for (int m = 0; m < 4; ++m)
        #pragma unroll
        for (int t = 0; t < 2; ++t)
            #pragma unroll
            for (int v = 0; v < 8; ++v) acc[m][t][v] = 0.0f;

    gemm_pass(hl, w1b, wave, l15, lhi, acc);

    float bias1[2];
    bias1[0] = b1[wave * 32 + l15];
    bias1[1] = b1[wave * 32 + 16 + l15];

#if HAVE_ASYNC_LDS
    __builtin_amdgcn_s_wait_asynccnt(0);   // sids landed (fenced by barrier below)
#endif
    __syncthreads();   // all waves done reading x tile -> safe to overwrite with h

    #pragma unroll
    for (int m = 0; m < 4; ++m)
        #pragma unroll
        for (int t = 0; t < 2; ++t) {
            int col = wave * 32 + t * 16 + l15;
            #pragma unroll
            for (int v = 0; v < 8; ++v) {
                float hv = acc[m][t][v] + bias1[t];
                hv = hv > 0.0f ? hv : 0.0f;
                acc[m][t][v] = hv;                                 // keep h in regs
                hl[(m * 16 + v + 8 * lhi) * HSTR + col] = (bf16_t)hv;
            }
        }
    __syncthreads();   // full h tile visible to all waves

    // ---------------- GEMM2: att = sigmoid(h @ W2^T + b2) ----------------
    v8f acc2[4][2];
    #pragma unroll
    for (int m = 0; m < 4; ++m)
        #pragma unroll
        for (int t = 0; t < 2; ++t)
            #pragma unroll
            for (int v = 0; v < 8; ++v) acc2[m][t][v] = 0.0f;

    gemm_pass(hl, w2b, wave, l15, lhi, acc2);

    float bias2[2];
    bias2[0] = b2[wave * 32 + l15];
    bias2[1] = b2[wave * 32 + 16 + l15];

    // g = h * sigmoid(z); g >= 0 since h >= 0, att in (0,1)
    #pragma unroll
    for (int m = 0; m < 4; ++m)
        #pragma unroll
        for (int t = 0; t < 2; ++t)
            #pragma unroll
            for (int v = 0; v < 8; ++v) {
                float z   = acc2[m][t][v] + bias2[t];
                float att = 1.0f / (1.0f + __expf(-z));
                acc2[m][t][v] = acc[m][t][v] * att;   // g
            }

    // ---------------- segmented reduction (batch is sorted) ----------------
    if (tid == 0) {
        int ra = 0;
        while (ra < TM) {
            int s  = sids[ra];
            int rb = ra + 1;
            while (rb < TM && sids[rb] == s) rb++;
            atomicAdd(&cnt[s], (float)(rb - ra));
            ra = rb;
        }
    }

    int ra = 0;
    while (ra < TM) {
        int s  = sids[ra];
        int rb = ra + 1;
        while (rb < TM && sids[rb] == s) rb++;

        #pragma unroll
        for (int t = 0; t < 2; ++t) {
            float sum = 0.0f, mx = 0.0f;
            #pragma unroll
            for (int m = 0; m < 4; ++m)
                #pragma unroll
                for (int v = 0; v < 8; ++v) {
                    int  row = m * 16 + v + 8 * lhi;
                    bool in  = (row >= ra) && (row < rb);
                    float gv = acc2[m][t][v];
                    sum += in ? gv : 0.0f;
                    mx   = in ? fmaxf(mx, gv) : mx;
                }
            // fold lanes 16..31 (rows +8) onto lanes 0..15 (wave32 C layout)
            sum += __shfl_xor(sum, 16);
            float om = __shfl_xor(mx, 16);
            mx = fmaxf(mx, om);
            if (lane < 16) {
                int col = wave * 32 + t * 16 + lane;
                atomicAdd(&out[(size_t)s * (2 * H) + H + col], sum);
                // g >= 0 so IEEE bit pattern is monotonic under unsigned max
                atomicMax((unsigned int*)&out[(size_t)s * (2 * H) + col],
                          __float_as_uint(mx));
            }
        }
        ra = rb;
    }
}

// ---------------------------------------------------------------------------
// finalize: mean = sum / count (zero for empty segments)
// ---------------------------------------------------------------------------
__global__ __launch_bounds__(256) void wr_finalize_kernel(
    float* __restrict__ out, const float* __restrict__ cnt)
{
    int i = blockIdx.x * 256 + threadIdx.x;
    if (i >= NSEG * H) return;
    int b = i >> 8;
    int c = i & 255;
    float n = cnt[b];
    float* p = &out[(size_t)b * (2 * H) + H + c];
    *p = (n > 0.0f) ? (*p / n) : 0.0f;
}

// ---------------------------------------------------------------------------
extern "C" void kernel_launch(void* const* d_in, const int* in_sizes, int n_in,
                              void* d_out, int out_size, void* d_ws, size_t ws_size,
                              hipStream_t stream)
{
    const float* x     = (const float*)d_in[0];
    const int*   batch = (const int*)d_in[1];
    const float* W1    = (const float*)d_in[2];
    const float* b1    = (const float*)d_in[3];
    const float* W2    = (const float*)d_in[4];
    const float* b2    = (const float*)d_in[5];
    float* out = (float*)d_out;

    bf16_t* w1b = (bf16_t*)d_ws;
    bf16_t* w2b = w1b + (size_t)H * H;
    float*  cnt = (float*)(w2b + (size_t)H * H);

    const int N = in_sizes[0] / H;   // 200000 -> 3125 tiles of 64 rows

    wr_init_kernel<<<(NSEG * 2 * H + 255) / 256, 256, 0, stream>>>(
        W1, W2, w1b, w2b, cnt, out);
    wr_fused_kernel<<<N / TM, 256, 0, stream>>>(
        x, batch, w1b, b1, w2b, b2, out, cnt);
    wr_finalize_kernel<<<(NSEG * H + 255) / 256, 256, 0, stream>>>(out, cnt);
}
